// Actor_Soft_Attention_73306501808520
// MI455X (gfx1250) — matrix operand order repaired and checked by reading, not compiled, not run on hardware
//
#include <hip/hip_runtime.h>
#include <hip/hip_bf16.h>
#include <math.h>

// ---------------------------------------------------------------------------
// Actor soft-attention MLP, fused, bf16 WMMA (gfx1250 / CDNA5, wave32)
//   B=65536, IN=128, HID=256, OUT=8
// One wave per block; each wave processes 32 batch rows (two 16-row M-tiles)
// through all 4 GEMMs -> 2x weight-fragment reuse vs 16-row version and
// 8 independent WMMA accumulators per k-step for ILP.
// ---------------------------------------------------------------------------

typedef __attribute__((ext_vector_type(16))) __bf16 v16bf;
typedef __attribute__((ext_vector_type(8)))  __bf16 v8bf;
typedef __attribute__((ext_vector_type(4)))  __bf16 v4bf;
typedef __attribute__((ext_vector_type(8)))  float  v8f;

#define B_TOTAL 65536
#define HID     256
#define OUT_F   8

#define TPB            32                // one wave per block
#define ROWS_PER_BLOCK 32                // two 16-row M-tiles

// xrow: 32 rows x 640 cols (cols [0,256)=h_i, cols [256,640)=bf16 s0|s1|s2)
// stride 648 bf16 -> 1296 B -> 324 dwords; 324 % 64 = 4 banks/row step:
// 16 consecutive rows hit 16 distinct bank quads for b128 loads.
#define XSTRIDE  648
#define H3STRIDE 264                     // 256 + 8 pad, same trick
#define LDS_ELEMS (32 * XSTRIDE + 32 * H3STRIDE)   // 29184 bf16 = 58368 B

// workspace layout (bf16 elements)
#define W1_OFF 0
#define W2_OFF 65536
#define W3_OFF 131072
#define W4_OFF 294912                    // W4 zero-padded to 16 x 256
#define WS_ELEMS 299008

// --------------------------- helpers ---------------------------------------

static __device__ inline v8f wmma_bf16(v16bf a, v16bf b, v8f c) {
    // (neg_a, A, neg_b, B, c_mod, C, reuse_a, reuse_b)
    return __builtin_amdgcn_wmma_f32_16x16x32_bf16(
        false, a, false, b, (short)0, c, false, false);
}

// A-fragment (16x32 bf16): lane half h, row m.  p = &row[colbase + h*8].
// chunk0 = cols [colbase+h*8, +8), chunk1 = cols [colbase+16+h*8, +8).
static __device__ inline v16bf load_a_frag(const __bf16* p) {
    v8bf lo = *(const v8bf*)p;
    v8bf hi = *(const v8bf*)(p + 16);
    return __builtin_shufflevector(lo, hi,
        0,1,2,3,4,5,6,7,8,9,10,11,12,13,14,15);
}

// B-fragment (32x16 bf16): lane half h, weight row n.  p = &row[colbase + h*8].
// chunk0 = K [colbase+h*8, +8), chunk1 = K [colbase+16+h*8, +8).
static __device__ inline v16bf load_b_frag(const __bf16* p) {
    v8bf lo = *(const v8bf*)p;
    v8bf hi = *(const v8bf*)(p + 16);
    return __builtin_shufflevector(lo, hi,
        0,1,2,3,4,5,6,7,8,9,10,11,12,13,14,15);
}

#define VZERO {0,0,0,0,0,0,0,0}

// --------------------------- prep: f32 weights -> bf16 ----------------------

__global__ __launch_bounds__(256) void prep_weights(
    const float* __restrict__ W1, const float* __restrict__ W2,
    const float* __restrict__ W3, const float* __restrict__ W4,
    __bf16* __restrict__ ws)
{
    int i = blockIdx.x * blockDim.x + threadIdx.x;
    if (i >= WS_ELEMS) return;
    if (i < W2_OFF) {
        ws[i] = (__bf16)W1[i];
    } else if (i < W3_OFF) {
        ws[i] = (__bf16)W2[i - W2_OFF];
    } else if (i < W4_OFF) {
        ws[i] = (__bf16)W3[i - W3_OFF];
    } else {
        int j = i - W4_OFF;              // 16 x 256, rows >= 8 are zero pad
        int r = j >> 8;
        ws[i] = (r < OUT_F) ? (__bf16)W4[j] : (__bf16)0.0f;
    }
}

// --------------------------- fused main kernel ------------------------------

__global__ __launch_bounds__(TPB, 1) void actor_fused(
    const float*  __restrict__ state,   // (B, 3, 128)
    const __bf16* __restrict__ wsb,     // converted weights
    const float*  __restrict__ b1, const float* __restrict__ b2,
    const float*  __restrict__ b3, const float* __restrict__ b4,
    float* __restrict__ out)            // (B, 8)
{
    __shared__ __bf16 smem[LDS_ELEMS];

    const int lane = threadIdx.x & 31;
    const int half = lane >> 4;          // 0 or 1
    const int m    = lane & 15;          // row (A/C) or col (B/C) index
    const int rowbase = blockIdx.x * ROWS_PER_BLOCK;

    __bf16* xr = smem;                   // 32 x (640 + pad)
    __bf16* h3 = smem + 32 * XSTRIDE;    // 32 x (256 + pad)

    const __bf16* W1bf = wsb + W1_OFF;
    const __bf16* W2bf = wsb + W2_OFF;
    const __bf16* W3bf = wsb + W3_OFF;
    const __bf16* W4bf = wsb + W4_OFF;

    // ---- stage: s_cat = [s0|s1|s2] as bf16 into xr cols [256,640) ----------
    {
        const float* srow = state + (size_t)rowbase * 384;
        #pragma unroll 4
        for (int i = lane * 4; i < 32 * 384; i += 32 * 4) {
            float4 v = *(const float4*)(srow + i);
            int r = i / 384;
            int c = i - r * 384;
            v4bf pk = { (__bf16)v.x, (__bf16)v.y, (__bf16)v.z, (__bf16)v.w };
            *(v4bf*)(xr + r * XSTRIDE + 256 + c) = pk;
        }
    }
    __syncthreads();

    // ---- phase 1: GEMM1/GEMM2 + bias + relu + 2-way softmax -> h_i ---------
    // agentsA = [s0|s1] = s_cat cols [0,256) ; agentsB = [s0|s2] =
    // s_cat cols [0,128) U [256,384); for k<128 the two A-fragments coincide.
    const __bf16* arow0 = xr + m        * XSTRIDE + 256 + half * 8;  // tile 0
    const __bf16* arow1 = xr + (m + 16) * XSTRIDE + 256 + half * 8;  // tile 1
    #pragma unroll 1
    for (int nt = 0; nt < 16; ++nt) {
        const int h = nt * 16 + m;
        v8f hA0 = VZERO, hB0 = VZERO, eA0 = VZERO, eB0 = VZERO;
        v8f hA1 = VZERO, hB1 = VZERO, eA1 = VZERO, eB1 = VZERO;
        const __bf16* w1p = W1bf + h * 256 + half * 8;
        const __bf16* w2p = W2bf + h * 256 + half * 8;
        #pragma unroll
        for (int ks = 0; ks < 8; ++ks) {
            const int k0 = ks * 32;
            v16bf aA0 = load_a_frag(arow0 + k0);
            v16bf aA1 = load_a_frag(arow1 + k0);
            v16bf aB0 = (k0 < 128) ? aA0 : load_a_frag(arow0 + k0 + 128);
            v16bf aB1 = (k0 < 128) ? aA1 : load_a_frag(arow1 + k0 + 128);
            v16bf w1  = load_b_frag(w1p + k0);
            v16bf w2  = load_b_frag(w2p + k0);
            hA0 = wmma_bf16(aA0, w1, hA0);
            hA1 = wmma_bf16(aA1, w1, hA1);
            hB0 = wmma_bf16(aB0, w1, hB0);
            hB1 = wmma_bf16(aB1, w1, hB1);
            eA0 = wmma_bf16(aA0, w2, eA0);
            eA1 = wmma_bf16(aA1, w2, eA1);
            eB0 = wmma_bf16(aB0, w2, eB0);
            eB1 = wmma_bf16(aB1, w2, eB1);
        }
        const float bb1 = b1[h];
        const float bb2 = b2[h];
        #pragma unroll
        for (int g = 0; g < 8; ++g) {
            int M = g + half * 8;                       // C-frag row in tile
            // tile 0
            {
                float x0 = fmaxf(hA0[g] + bb1, 0.0f);
                float x1 = fmaxf(hB0[g] + bb1, 0.0f);
                float y0 = fmaxf(eA0[g] + bb2, 0.0f);
                float y1 = fmaxf(eB0[g] + bb2, 0.0f);
                float a0 = 1.0f / (1.0f + __expf(y1 - y0));
                float hi = x1 + a0 * (x0 - x1);
                xr[M * XSTRIDE + nt * 16 + m] = (__bf16)hi;
            }
            // tile 1
            {
                float x0 = fmaxf(hA1[g] + bb1, 0.0f);
                float x1 = fmaxf(hB1[g] + bb1, 0.0f);
                float y0 = fmaxf(eA1[g] + bb2, 0.0f);
                float y1 = fmaxf(eB1[g] + bb2, 0.0f);
                float a0 = 1.0f / (1.0f + __expf(y1 - y0));
                float hi = x1 + a0 * (x0 - x1);
                xr[(M + 16) * XSTRIDE + nt * 16 + m] = (__bf16)hi;
            }
        }
    }
    __syncthreads();

    // ---- phase 2: GEMM3 over concat(h_i, s0, s1, s2) (K=640) + relu --------
    const __bf16* a3row0 = xr + m        * XSTRIDE + half * 8;
    const __bf16* a3row1 = xr + (m + 16) * XSTRIDE + half * 8;
    #pragma unroll 1
    for (int nt = 0; nt < 16; ++nt) {
        const int h = nt * 16 + m;
        v8f acc0 = VZERO, acc1 = VZERO;
        const __bf16* wp = W3bf + h * 640 + half * 8;
        #pragma unroll 4
        for (int ks = 0; ks < 20; ++ks) {
            const int k0 = ks * 32;
            v16bf a0 = load_a_frag(a3row0 + k0);
            v16bf a1 = load_a_frag(a3row1 + k0);
            v16bf w  = load_b_frag(wp + k0);
            acc0 = wmma_bf16(a0, w, acc0);
            acc1 = wmma_bf16(a1, w, acc1);
        }
        const float bb = b3[h];
        #pragma unroll
        for (int g = 0; g < 8; ++g) {
            int M = g + half * 8;
            h3[M        * H3STRIDE + nt * 16 + m] = (__bf16)fmaxf(acc0[g] + bb, 0.0f);
            h3[(M + 16) * H3STRIDE + nt * 16 + m] = (__bf16)fmaxf(acc1[g] + bb, 0.0f);
        }
    }
    __syncthreads();

    // ---- phase 3: GEMM4 (8 outputs, padded to 16) + tanh -------------------
    {
        v8f acc0 = VZERO, acc1 = VZERO;
        const __bf16* wp  = W4bf + m * 256 + half * 8;  // padded rows safe
        const __bf16* ar0 = h3 + m        * H3STRIDE + half * 8;
        const __bf16* ar1 = h3 + (m + 16) * H3STRIDE + half * 8;
        #pragma unroll
        for (int ks = 0; ks < 8; ++ks) {
            v16bf a0 = load_a_frag(ar0 + ks * 32);
            v16bf a1 = load_a_frag(ar1 + ks * 32);
            v16bf w  = load_b_frag(wp + ks * 32);
            acc0 = wmma_bf16(a0, w, acc0);
            acc1 = wmma_bf16(a1, w, acc1);
        }
        if (m < OUT_F) {
            const float bb = b4[m];
            #pragma unroll
            for (int g = 0; g < 8; ++g) {
                int M = g + half * 8;
                out[((size_t)(rowbase + M))      * OUT_F + m] = tanhf(acc0[g] + bb);
                out[((size_t)(rowbase + M + 16)) * OUT_F + m] = tanhf(acc1[g] + bb);
            }
        }
    }
}

// --------------------------- launch -----------------------------------------

extern "C" void kernel_launch(void* const* d_in, const int* in_sizes, int n_in,
                              void* d_out, int out_size, void* d_ws, size_t ws_size,
                              hipStream_t stream) {
    (void)in_sizes; (void)n_in; (void)out_size; (void)ws_size;

    const float* state = (const float*)d_in[0];
    const float* W1    = (const float*)d_in[1];
    const float* b1    = (const float*)d_in[2];
    const float* W2    = (const float*)d_in[3];
    const float* b2    = (const float*)d_in[4];
    const float* W3    = (const float*)d_in[5];
    const float* b3    = (const float*)d_in[6];
    const float* W4    = (const float*)d_in[7];
    const float* b4    = (const float*)d_in[8];
    float*       out   = (float*)d_out;
    __bf16*      wsb   = (__bf16*)d_ws;

    // 1) convert weights to bf16 (W4 zero-padded to 16 rows)
    prep_weights<<<(WS_ELEMS + 255) / 256, 256, 0, stream>>>(W1, W2, W3, W4, wsb);

    // 2) fused attention MLP: 1 wave/block, 32 rows/wave
    actor_fused<<<B_TOTAL / ROWS_PER_BLOCK, TPB, 0, stream>>>(
        state, wsb, b1, b2, b3, b4, out);
}
